// MultiheadAttentionWithKVCache_68839735820824
// MI455X (gfx1250) — matrix operand order repaired and checked by reading, not compiled, not run on hardware
//
#include <hip/hip_runtime.h>
#include <hip/hip_bf16.h>
#include <math.h>

// ---- problem constants (match reference) ----
#define B_   2
#define S_   2048
#define E_   1024
#define H_   16
#define DH_  64
#define ROT_ 32
// log2(10000)/16
#define NEG_L2T_OVER_16 0.83048202372184065f

typedef __attribute__((ext_vector_type(16))) _Float16 v16h;
typedef __attribute__((ext_vector_type(8)))  _Float16 v8h;
typedef __attribute__((ext_vector_type(8)))  float    v8f;
typedef __attribute__((ext_vector_type(4)))  float    v4f;

union AFrag { v16h v; v8h h[2]; };

static __device__ inline v8f wmma16(v16h a, v16h b, v8f c) {
  // D = A(16x32 f16) * B(32x16 f16) + C(16x16 f32)
  return __builtin_amdgcn_wmma_f32_16x16x32_f16(false, a, false, b, (short)0, c,
                                                false, false);
}

static __device__ inline float rowMax16(float x) {
  x = fmaxf(x, __shfl_xor(x, 1, 32));
  x = fmaxf(x, __shfl_xor(x, 2, 32));
  x = fmaxf(x, __shfl_xor(x, 4, 32));
  x = fmaxf(x, __shfl_xor(x, 8, 32));
  return x;
}
static __device__ inline float rowSum16(float x) {
  x += __shfl_xor(x, 1, 32);
  x += __shfl_xor(x, 2, 32);
  x += __shfl_xor(x, 4, 32);
  x += __shfl_xor(x, 8, 32);
  return x;
}

// ============================================================================
// Projection GEMM: dst = rope_or_copy( X(4096xE) @ W^T + b ), emitted as f16.
// mode 0: Q (RoPE, layout (B,H,S,DH))
// mode 1: K (RoPE, layout (B,H,S,DH))
// mode 2: V (no RoPE, transposed layout (B,H,DH,S))
// Block: 128 thr (4 waves), tile 64x64, K-step 32, wave = 32x32 (2x2 WMMA).
// ============================================================================
__global__ __launch_bounds__(128)
void proj_rope_kernel(const float* __restrict__ X, const float* __restrict__ W,
                      const float* __restrict__ bias, _Float16* __restrict__ dst,
                      int mode)
{
  __shared__ __align__(16) _Float16 lA[64 * 32];
  __shared__ __align__(16) _Float16 lB[64 * 32];

  const int tid  = threadIdx.x & 127;   // provably < 128: kills EXEC guards
  const int lane = tid & 31;
  const int warp = tid >> 5;
  const int lr   = lane & 15;   // N / M-within-tile
  const int g    = lane >> 4;   // half-wave group
  const int bm   = blockIdx.x & 63;
  const int bn   = blockIdx.x >> 6;
  const int m0   = bm * 64;
  const int n0   = bn * 64;
  const int wM   = (warp >> 1) * 32;
  const int wN   = (warp & 1) * 32;

  v8f acc[2][2];
  #pragma unroll
  for (int i = 0; i < 2; ++i)
    #pragma unroll
    for (int j = 0; j < 2; ++j) acc[i][j] = (v8f)0.0f;

  for (int k0 = 0; k0 < E_; k0 += 32) {
    // stage A (rows of X) and B (rows of W == cols of W^T), f32 -> f16
    #pragma unroll
    for (int it = 0; it < 4; ++it) {
      const int sIdx = tid + it * 128;           // 0..511, no guard needed
      const int row  = sIdx >> 3;
      const int c4   = (sIdx & 7) * 4;
      v4f a = *(const v4f*)&X[(size_t)(m0 + row) * E_ + k0 + c4];
      lA[row * 32 + c4 + 0] = (_Float16)a.x;
      lA[row * 32 + c4 + 1] = (_Float16)a.y;
      lA[row * 32 + c4 + 2] = (_Float16)a.z;
      lA[row * 32 + c4 + 3] = (_Float16)a.w;
      v4f b = *(const v4f*)&W[(size_t)(n0 + row) * E_ + k0 + c4];
      lB[row * 32 + c4 + 0] = (_Float16)b.x;
      lB[row * 32 + c4 + 1] = (_Float16)b.y;
      lB[row * 32 + c4 + 2] = (_Float16)b.z;
      lB[row * 32 + c4 + 3] = (_Float16)b.w;
    }
    __syncthreads();

    AFrag a[2], bf[2];
    #pragma unroll
    for (int i = 0; i < 2; ++i) {
      const int row = wM + i * 16 + lr;
      a[i].h[0] = *(const v8h*)&lA[row * 32 + g * 8];
      a[i].h[1] = *(const v8h*)&lA[row * 32 + 16 + g * 8];
    }
    #pragma unroll
    for (int j = 0; j < 2; ++j) {
      const int n = wN + j * 16 + lr;
      bf[j].h[0] = *(const v8h*)&lB[n * 32 + 8 * g];
      bf[j].h[1] = *(const v8h*)&lB[n * 32 + 16 + 8 * g];
    }
    #pragma unroll
    for (int i = 0; i < 2; ++i)
      #pragma unroll
      for (int j = 0; j < 2; ++j)
        acc[i][j] = wmma16(a[i].v, bf[j].v, acc[i][j]);
    __syncthreads();
  }

  // epilogue: bias (+ RoPE for Q/K), scatter to head layout as f16
  #pragma unroll
  for (int i = 0; i < 2; ++i) {
    #pragma unroll
    for (int j = 0; j < 2; ++j) {
      #pragma unroll
      for (int r = 0; r < 8; ++r) {
        const int row = m0 + wM + i * 16 + r + 8 * g; // (b*S + s)
        const int col = n0 + wN + j * 16 + lr;        // (h*DH + d)
        float x = acc[i][j][r] + bias[col];
        const int s = row & (S_ - 1);
        const int b = row >> 11;
        const int h = col >> 6;
        const int d = col & (DH_ - 1);
        // partner element for RoPE pair lives in lane^1 (col^1, same row)
        const float partner = __shfl_xor(x, 1, 32);
        if (mode <= 1) {
          float y = x;
          if (d < ROT_) {
            const int   p   = d >> 1;
            // theta^(-p/16) == exp2(-p * log2(theta)/16): one v_exp_f32
            const float inv_freq = exp2f(-(float)p * NEG_L2T_OVER_16);
            const float ang = (float)s * inv_freq;
            float sn, cs;
            __sincosf(ang, &sn, &cs);
            y = (d & 1) ? (x * cs + partner * sn) : (x * cs - partner * sn);
          }
          dst[(((size_t)(b * H_ + h)) * S_ + s) * DH_ + d] = (_Float16)y;
        } else {
          // V transposed: (B,H,DH,S) so PV B-fragments are contiguous in key
          dst[(((size_t)(b * H_ + h)) * DH_ + d) * S_ + s] = (_Float16)x;
        }
      }
    }
  }
}

// ============================================================================
// Flash attention: one wave per 16 query rows of one (b,h); key blocks of 32.
// scores via 2x2 WMMA, online softmax (16-lane shfl reductions), P through
// LDS to convert C-layout -> A-layout, PV via 4 WMMAs into 16x64 f32 acc.
// ============================================================================
__global__ __launch_bounds__(128)
void flash_attn_kernel(const _Float16* __restrict__ Qh,
                       const _Float16* __restrict__ Kh,
                       const _Float16* __restrict__ Vt,
                       _Float16* __restrict__ Oh)
{
  __shared__ __align__(16) _Float16 ldsP[4][16 * 32];

  const int tid  = threadIdx.x & 127;
  const int lane = tid & 31;
  const int warp = tid >> 5;
  const int lr   = lane & 15;
  const int g    = lane >> 4;

  const int wid   = blockIdx.x * 4 + warp;   // 4096 waves total
  const int qTile = wid & 127;               // S/16
  const int bh    = wid >> 7;                // 0..B*H-1
  const int q0    = qTile * 16;
  const int b     = bh >> 4;
  const int h     = bh & 15;

  // Q A-fragments (d 0..31 and 32..63)
  const _Float16* qb = Qh + ((size_t)bh * S_ + q0 + lr) * DH_;
  AFrag A0, A1;
  A0.h[0] = *(const v8h*)(qb + g * 8);
  A0.h[1] = *(const v8h*)(qb + 16 + g * 8);
  A1.h[0] = *(const v8h*)(qb + 32 + g * 8);
  A1.h[1] = *(const v8h*)(qb + 48 + g * 8);

  v8f O[4];
  #pragma unroll
  for (int i = 0; i < 4; ++i) O[i] = (v8f)0.0f;
  float m[8], l[8];
  #pragma unroll
  for (int r = 0; r < 8; ++r) { m[r] = -1e30f; l[r] = 0.0f; }

  _Float16* P = ldsP[warp];
  const float scale = 0.125f; // 1/sqrt(64)

  for (int j0 = 0; j0 <= q0 + 15; j0 += 32) {
    // K^T B-fragments: key tile 0 (j0..j0+15) and tile 1 (j0+16..j0+31)
    const _Float16* kb0 = Kh + ((size_t)bh * S_ + j0 + lr) * DH_;
    const _Float16* kb1 = kb0 + 16 * DH_;
    AFrag B0a, B0b, B1a, B1b;
    B0a.h[0] = *(const v8h*)(kb0 + 8 * g);      B0a.h[1] = *(const v8h*)(kb0 + 16 + 8 * g);
    B0b.h[0] = *(const v8h*)(kb0 + 32 + 8 * g); B0b.h[1] = *(const v8h*)(kb0 + 48 + 8 * g);
    B1a.h[0] = *(const v8h*)(kb1 + 8 * g);      B1a.h[1] = *(const v8h*)(kb1 + 16 + 8 * g);
    B1b.h[0] = *(const v8h*)(kb1 + 32 + 8 * g); B1b.h[1] = *(const v8h*)(kb1 + 48 + 8 * g);

    v8f s0 = (v8f)0.0f, s1 = (v8f)0.0f;
    s0 = wmma16(A0.v, B0a.v, s0);
    s0 = wmma16(A1.v, B0b.v, s0);
    s1 = wmma16(A0.v, B1a.v, s1);
    s1 = wmma16(A1.v, B1b.v, s1);

    #pragma unroll
    for (int r = 0; r < 8; ++r) {
      const int qrow = q0 + r + 8 * g;
      float x0 = s0[r] * scale; if (j0 + lr      > qrow) x0 = -1e30f;
      float x1 = s1[r] * scale; if (j0 + 16 + lr > qrow) x1 = -1e30f;
      float mt = rowMax16(fmaxf(x0, x1));
      const float mn = fmaxf(m[r], mt);
      const float sc = __expf(m[r] - mn);
      const float e0 = __expf(x0 - mn);
      const float e1 = __expf(x1 - mn);
      l[r] = l[r] * sc + rowSum16(e0 + e1);
      m[r] = mn;
      O[0][r] *= sc; O[1][r] *= sc; O[2][r] *= sc; O[3][r] *= sc;
      P[(r + 8 * g) * 32 + lr]      = (_Float16)e0;
      P[(r + 8 * g) * 32 + 16 + lr] = (_Float16)e1;
    }
    asm volatile("s_wait_dscnt 0" ::: "memory");

    AFrag Pf;
    Pf.h[0] = *(const v8h*)&P[lr * 32 + g * 8];
    Pf.h[1] = *(const v8h*)&P[lr * 32 + 16 + g * 8];

    #pragma unroll
    for (int tt = 0; tt < 4; ++tt) {
      const _Float16* vb = Vt + ((size_t)bh * DH_ + tt * 16 + lr) * S_ + j0 + 8 * g;
      AFrag Vf;
      Vf.h[0] = *(const v8h*)vb;
      Vf.h[1] = *(const v8h*)(vb + 16);
      O[tt] = wmma16(Pf.v, Vf.v, O[tt]);
    }
  }

  // normalize and write attention output in (B,S,E) f16 for the out-proj
  #pragma unroll
  for (int r = 0; r < 8; ++r) {
    const float inv = 1.0f / l[r];
    const int   s   = q0 + r + 8 * g;
    const size_t base = ((size_t)b * S_ + s) * E_ + h * DH_ + lr;
    Oh[base + 0]  = (_Float16)(O[0][r] * inv);
    Oh[base + 16] = (_Float16)(O[1][r] * inv);
    Oh[base + 32] = (_Float16)(O[2][r] * inv);
    Oh[base + 48] = (_Float16)(O[3][r] * inv);
  }
}

// ============================================================================
// Output projection: out(f32) = Oh(f16, 4096xE) @ Wo^T + bo
// ============================================================================
__global__ __launch_bounds__(128)
void oproj_kernel(const _Float16* __restrict__ Xh, const float* __restrict__ W,
                  const float* __restrict__ bias, float* __restrict__ out)
{
  __shared__ __align__(16) _Float16 lA[64 * 32];
  __shared__ __align__(16) _Float16 lB[64 * 32];

  const int tid  = threadIdx.x & 127;
  const int lane = tid & 31;
  const int warp = tid >> 5;
  const int lr   = lane & 15;
  const int g    = lane >> 4;
  const int bm   = blockIdx.x & 63;
  const int bn   = blockIdx.x >> 6;
  const int m0   = bm * 64;
  const int n0   = bn * 64;
  const int wM   = (warp >> 1) * 32;
  const int wN   = (warp & 1) * 32;

  v8f acc[2][2];
  #pragma unroll
  for (int i = 0; i < 2; ++i)
    #pragma unroll
    for (int j = 0; j < 2; ++j) acc[i][j] = (v8f)0.0f;

  for (int k0 = 0; k0 < E_; k0 += 32) {
    #pragma unroll
    for (int it = 0; it < 2; ++it) {            // A is already f16: 8-half chunks
      const int sIdx = tid + it * 128;          // 0..255
      const int row  = sIdx >> 2;
      const int c8   = (sIdx & 3) * 8;
      *(v8h*)&lA[row * 32 + c8] =
          *(const v8h*)&Xh[(size_t)(m0 + row) * E_ + k0 + c8];
    }
    #pragma unroll
    for (int it = 0; it < 4; ++it) {            // B: f32 -> f16
      const int sIdx = tid + it * 128;          // 0..511
      const int row  = sIdx >> 3;
      const int c4   = (sIdx & 7) * 4;
      v4f bv = *(const v4f*)&W[(size_t)(n0 + row) * E_ + k0 + c4];
      lB[row * 32 + c4 + 0] = (_Float16)bv.x;
      lB[row * 32 + c4 + 1] = (_Float16)bv.y;
      lB[row * 32 + c4 + 2] = (_Float16)bv.z;
      lB[row * 32 + c4 + 3] = (_Float16)bv.w;
    }
    __syncthreads();

    AFrag a[2], bf[2];
    #pragma unroll
    for (int i = 0; i < 2; ++i) {
      const int row = wM + i * 16 + lr;
      a[i].h[0] = *(const v8h*)&lA[row * 32 + g * 8];
      a[i].h[1] = *(const v8h*)&lA[row * 32 + 16 + g * 8];
    }
    #pragma unroll
    for (int j = 0; j < 2; ++j) {
      const int n = wN + j * 16 + lr;
      bf[j].h[0] = *(const v8h*)&lB[n * 32 + 8 * g];
      bf[j].h[1] = *(const v8h*)&lB[n * 32 + 16 + 8 * g];
    }
    #pragma unroll
    for (int i = 0; i < 2; ++i)
      #pragma unroll
      for (int j = 0; j < 2; ++j)
        acc[i][j] = wmma16(a[i].v, bf[j].v, acc[i][j]);
    __syncthreads();
  }

  #pragma unroll
  for (int i = 0; i < 2; ++i)
    #pragma unroll
    for (int j = 0; j < 2; ++j)
      #pragma unroll
      for (int r = 0; r < 8; ++r) {
        const int row = m0 + wM + i * 16 + r + 8 * g;
        const int col = n0 + wN + j * 16 + lr;
        out[(size_t)row * E_ + col] = acc[i][j][r] + bias[col];
      }
}

// ============================================================================
extern "C" void kernel_launch(void* const* d_in, const int* in_sizes, int n_in,
                              void* d_out, int out_size, void* d_ws, size_t ws_size,
                              hipStream_t stream) {
  (void)in_sizes; (void)n_in; (void)out_size; (void)ws_size;
  const float* query = (const float*)d_in[0];
  const float* key   = (const float*)d_in[1];
  const float* value = (const float*)d_in[2];
  const float* Wq    = (const float*)d_in[3];
  const float* bq    = (const float*)d_in[4];
  const float* Wk    = (const float*)d_in[5];
  const float* bk    = (const float*)d_in[6];
  const float* Wv    = (const float*)d_in[7];
  const float* bv    = (const float*)d_in[8];
  const float* Wo    = (const float*)d_in[9];
  const float* bo    = (const float*)d_in[10];
  float* out = (float*)d_out;

  _Float16* ws = (_Float16*)d_ws;
  const size_t N = (size_t)B_ * H_ * S_ * DH_;  // 4,194,304 halves per tensor
  _Float16* Qh = ws;
  _Float16* Kh = ws + N;
  _Float16* Vt = ws + 2 * N;
  _Float16* Oh = ws + 3 * N;   // total 32 MB of f16 workspace

  const dim3 blk(128);
  const dim3 grid(1024);
  proj_rope_kernel<<<grid, blk, 0, stream>>>(query, Wq, bq, Qh, 0);
  proj_rope_kernel<<<grid, blk, 0, stream>>>(key,   Wk, bk, Kh, 1);
  proj_rope_kernel<<<grid, blk, 0, stream>>>(value, Wv, bv, Vt, 2);
  flash_attn_kernel<<<grid, blk, 0, stream>>>(Qh, Kh, Vt, Oh);
  oproj_kernel<<<grid, blk, 0, stream>>>(Oh, Wo, bo, out);
}
